// Net_15204184227887
// MI455X (gfx1250) — compile-verified, matrix operand order checked by reading, not run on hardware
//
#include <hip/hip_runtime.h>
#include <math.h>

// ---------------- problem constants ----------------
#define BB   64
#define NEXO 3
#define NI   (BB*NEXO)        // 192 exo images
#define DD   384
#define D4   1536
#define HH   14
#define WW   14
#define HWP  196              // Hh*Ww
#define NCLS 36
#define KK   3
#define NPTS (NEXO*HWP)       // 588
#define TOKSTRIDE 197         // HW+1 tokens per image in *_key

typedef __attribute__((ext_vector_type(16))) _Float16 v16h;
typedef __attribute__((ext_vector_type(8)))  float    v8f;

// ---------------- WMMA helpers ----------------
__device__ __forceinline__ v8f zero_v8f() {
  v8f z;
#pragma unroll
  for (int i = 0; i < 8; ++i) z[i] = 0.f;
  return z;
}

__device__ __forceinline__ v8f wmma_f16(v16h a, v16h b, v8f c) {
  return __builtin_amdgcn_wmma_f32_16x16x32_f16(false, a, false, b, (short)0, c, false, false);
}

// 16-bit A/B fragment loader (16x32 A layout, mirrored for 32x16 B).
// rowp points at the row (A: M=lane&15) / column-line (B: N=lane&15) base.
__device__ __forceinline__ v16h load_frag(const _Float16* rowp, int k0, int lane, bool valid) {
  v16h a;
  const int hi = (lane & 16) ? 8 : 0;
#pragma unroll
  for (int j = 0; j < 8; ++j) {
    int kk = k0 + ((j < 4) ? (hi + 2 * j) : (16 + hi + 2 * (j - 4)));
    _Float16 x0 = (_Float16)0.f, x1 = (_Float16)0.f;
    if (valid) { x0 = rowp[kk]; x1 = rowp[kk + 1]; }
    a[2 * j]     = x0;
    a[2 * j + 1] = x1;
  }
  return a;
}

// ---------------- block reductions (blockDim == 256) ----------------
__device__ __forceinline__ float block_sum(float v, float* red) {
  int tid = threadIdx.x;
  red[tid] = v; __syncthreads();
  for (int s = 128; s > 0; s >>= 1) { if (tid < s) red[tid] += red[tid + s]; __syncthreads(); }
  float r = red[0]; __syncthreads();
  return r;
}
__device__ __forceinline__ float block_max(float v, float* red) {
  int tid = threadIdx.x;
  red[tid] = v; __syncthreads();
  for (int s = 128; s > 0; s >>= 1) { if (tid < s) red[tid] = fmaxf(red[tid], red[tid + s]); __syncthreads(); }
  float r = red[0]; __syncthreads();
  return r;
}
__device__ __forceinline__ float block_min(float v, float* red) {
  int tid = threadIdx.x;
  red[tid] = v; __syncthreads();
  for (int s = 128; s > 0; s >>= 1) { if (tid < s) red[tid] = fminf(red[tid], red[tid + s]); __syncthreads(); }
  float r = red[0]; __syncthreads();
  return r;
}
__device__ __forceinline__ int block_argmax(float v, int idx, float* redf, int* redi) {
  int tid = threadIdx.x;
  redf[tid] = v; redi[tid] = idx; __syncthreads();
  for (int s = 128; s > 0; s >>= 1) {
    if (tid < s) {
      float a = redf[tid], b = redf[tid + s];
      if (b > a || (b == a && redi[tid + s] < redi[tid])) { redf[tid] = b; redi[tid] = redi[tid + s]; }
    }
    __syncthreads();
  }
  int r = redi[0]; __syncthreads();
  return r;
}

// ---------------- conversion kernels ----------------
__global__ void cvt_f16(const float* __restrict__ in, _Float16* __restrict__ out, int n) {
  int i = blockIdx.x * 256 + threadIdx.x;
  if (i < n) out[i] = (_Float16)in[i];
}
// (O,I,3,3) f32 -> [O][tap][I] f16 (tap-major so WMMA B loads are contiguous in Cin)
__global__ void cvt_convw(const float* __restrict__ in, _Float16* __restrict__ out) {
  int i = blockIdx.x * 256 + threadIdx.x;
  if (i >= DD * DD * 9) return;
  int o = i / (DD * 9), r = i % (DD * 9);
  int tap = r / DD, ci = r % DD;
  out[i] = (_Float16)in[(size_t)(o * DD + ci) * 9 + tap];
}

// ---------------- LayerNorm stats ----------------
__global__ void ln_stats(const float* __restrict__ key, float* __restrict__ out, int ntok) {
  int i = blockIdx.x * 256 + threadIdx.x;
  if (i >= ntok) return;
  int img = i / HWP, t = i % HWP;
  const float* row = key + ((size_t)(img * TOKSTRIDE + 1 + t)) * DD;
  float s = 0.f, s2 = 0.f;
  for (int c = 0; c < DD; ++c) { float x = row[c]; s += x; s2 += x * x; }
  float mu = s / (float)DD;
  float var = s2 / (float)DD - mu * mu;
  out[i] = mu;
  out[ntok + i] = rsqrtf(var + 1e-5f);
}

// ---------------- fused token MLP (LN -> GEMM1 -> GELU -> GEMM2 -> +residual) ----------------
__global__ __launch_bounds__(256) void mlp_fused(
    const float* __restrict__ key, const float* __restrict__ ln_g, const float* __restrict__ ln_b,
    const float* __restrict__ mustd,
    const _Float16* __restrict__ w1h, const float* __restrict__ b1,
    const _Float16* __restrict__ w2h, const float* __restrict__ b2,
    _Float16* __restrict__ out_hwc, int ntok) {
  __shared__ _Float16 lds_a[16 * DD];   // LN(x) f16, 12 KB
  __shared__ _Float16 lds_h[16 * D4];   // gelu(h1) f16, 48 KB
  const int tile = blockIdx.x, tid = threadIdx.x;
  const int lane = tid & 31, wave = tid >> 5;

  // stage LN(x) -> f16
  for (int i = tid; i < 16 * DD; i += 256) {
    int r = i / DD, c = i % DD;
    int token = tile * 16 + r;
    int img = token / HWP, t = token % HWP;
    float x = key[((size_t)(img * TOKSTRIDE + 1 + t)) * DD + c];
    float mu = mustd[token], rs = mustd[ntok + token];
    lds_a[i] = (_Float16)(((x - mu) * rs) * ln_g[c] + ln_b[c]);
  }
  __syncthreads();

  // GEMM1 (16 x 1536, K=384) + bias + exact GELU -> lds_h
  for (int nt = wave; nt < D4 / 16; nt += 8) {
    v8f acc = zero_v8f();
    const _Float16* arow = &lds_a[(lane & 15) * DD];
    const _Float16* brow = &w1h[(size_t)(nt * 16 + (lane & 15)) * DD];
    for (int ks = 0; ks < DD / 32; ++ks) {
      v16h a = load_frag(arow, ks * 32, lane, true);
      v16h b = load_frag(brow, ks * 32, lane, true);
      acc = wmma_f16(a, b, acc);
    }
    int n = nt * 16 + (lane & 15);
    float bias = b1[n];
#pragma unroll
    for (int r = 0; r < 8; ++r) {
      int m = r + ((lane & 16) ? 8 : 0);
      float v = acc[r] + bias;
      v = 0.5f * v * (1.0f + erff(v * 0.70710678118654752f));
      lds_h[m * D4 + n] = (_Float16)v;
    }
  }
  __syncthreads();

  // GEMM2 (16 x 384, K=1536) + bias + residual -> f16 HWC
  for (int nt = wave; nt < DD / 16; nt += 8) {
    v8f acc = zero_v8f();
    const _Float16* arow = &lds_h[(lane & 15) * D4];
    const _Float16* brow = &w2h[(size_t)(nt * 16 + (lane & 15)) * D4];
    for (int ks = 0; ks < D4 / 32; ++ks) {
      v16h a = load_frag(arow, ks * 32, lane, true);
      v16h b = load_frag(brow, ks * 32, lane, true);
      acc = wmma_f16(a, b, acc);
    }
    int n = nt * 16 + (lane & 15);
    float bias = b2[n];
#pragma unroll
    for (int r = 0; r < 8; ++r) {
      int m = r + ((lane & 16) ? 8 : 0);
      int token = tile * 16 + m;
      int img = token / HWP, t = token % HWP;
      float xin = key[((size_t)(img * TOKSTRIDE + 1 + t)) * DD + n];
      out_hwc[(size_t)token * DD + n] = (_Float16)(acc[r] + bias + xin);
    }
  }
}

// ---------------- 3x3 conv (SAME) as implicit GEMM via WMMA ----------------
// src: f16 HWC [nimg][196][384]; w: f16 [O][tap][I]; dst: f32 NCHW
__global__ __launch_bounds__(256) void conv3x3_wmma(
    const _Float16* __restrict__ src_hwc, const _Float16* __restrict__ wkh,
    const float* __restrict__ bias, float* __restrict__ dst) {
  const int img = blockIdx.y;
  const int tid = threadIdx.x, lane = tid & 31, wave = tid >> 5;
  const int tileId = blockIdx.x * 8 + wave;           // 13 M-tiles x 24 N-tiles = 312
  const int tm = tileId % 13, tn = tileId / 13;
  v8f acc = zero_v8f();
  const _Float16* srcimg = src_hwc + (size_t)img * HWP * DD;
  const int mrow = tm * 16 + (lane & 15);
  const int hh = mrow / WW, ww = mrow % WW;
  const bool mv = (mrow < HWP);
  const _Float16* wbase = wkh + (size_t)(tn * 16 + (lane & 15)) * (DD * 9);
#pragma unroll
  for (int tap = 0; tap < 9; ++tap) {
    int dy = tap / 3 - 1, dx = tap % 3 - 1;
    int h2 = hh + dy, w2 = ww + dx;
    bool valid = mv && (h2 >= 0) && (h2 < HH) && (w2 >= 0) && (w2 < WW);
    const _Float16* rowp = srcimg + (size_t)(h2 * WW + w2) * DD;
    const _Float16* wrow = wbase + (size_t)tap * DD;
    for (int ks = 0; ks < DD / 32; ++ks) {
      v16h a = load_frag(rowp, ks * 32, lane, valid);
      v16h b = load_frag(wrow, ks * 32, lane, true);
      acc = wmma_f16(a, b, acc);
    }
  }
  int n = tn * 16 + (lane & 15);
  float bv = bias[n];
  float* drow = dst + (size_t)img * DD * HWP + (size_t)n * HWP;
#pragma unroll
  for (int r = 0; r < 8; ++r) {
    int m = tm * 16 + r + ((lane & 16) ? 8 : 0);
    if (m < HWP) drow[m] = acc[r] + bv;
  }
}

// ---------------- BatchNorm stats / apply ----------------
__global__ __launch_bounds__(256) void bn_stats(const float* __restrict__ y, int nimg,
                                                float* __restrict__ mu, float* __restrict__ rstd) {
  __shared__ float red[256];
  int c = blockIdx.x, tid = threadIdx.x;
  int n = nimg * HWP;
  float s = 0.f, s2 = 0.f;
  for (int i = tid; i < n; i += 256) {
    int img = i / HWP, p = i % HWP;
    float x = y[(size_t)img * DD * HWP + (size_t)c * HWP + p];
    s += x; s2 += x * x;
  }
  float ts = block_sum(s, red);
  float ts2 = block_sum(s2, red);
  if (tid == 0) {
    float m = ts / (float)n;
    float v = ts2 / (float)n - m * m;
    mu[c] = m;
    rstd[c] = rsqrtf(v + 1e-5f);
  }
}

__global__ void bn_relu_f16hwc(const float* __restrict__ y, const float* __restrict__ mu,
                               const float* __restrict__ rstd, const float* __restrict__ g,
                               const float* __restrict__ bt, _Float16* __restrict__ out, int nimg) {
  size_t idx = (size_t)blockIdx.x * 256 + threadIdx.x;
  size_t tot = (size_t)nimg * DD * HWP;
  if (idx >= tot) return;
  size_t img = idx / ((size_t)DD * HWP);
  size_t rem = idx % ((size_t)DD * HWP);
  int c = (int)(rem / HWP), p = (int)(rem % HWP);
  float v = (y[idx] - mu[c]) * rstd[c] * g[c] + bt[c];
  v = fmaxf(v, 0.f);
  out[img * HWP * DD + (size_t)p * DD + c] = (_Float16)v;
}

__global__ void bn_relu_f32(const float* __restrict__ y, const float* __restrict__ mu,
                            const float* __restrict__ rstd, const float* __restrict__ g,
                            const float* __restrict__ bt, float* __restrict__ out, int nimg) {
  size_t idx = (size_t)blockIdx.x * 256 + threadIdx.x;
  size_t tot = (size_t)nimg * DD * HWP;
  if (idx >= tot) return;
  size_t rem = idx % ((size_t)DD * HWP);
  int c = (int)(rem / HWP);
  float v = (y[idx] - mu[c]) * rstd[c] * g[c] + bt[c];
  out[idx] = fmaxf(v, 0.f);
}

// ---------------- 1x1 conv (384 -> 36) + spatial-mean logits ----------------
__global__ __launch_bounds__(256) void fc_conv(const float* __restrict__ x, const float* __restrict__ w,
                                               const float* __restrict__ b, float* __restrict__ cam,
                                               float* __restrict__ logits) {
  __shared__ float red[256];
  int blk = blockIdx.x;
  int img = blk / NCLS, cls = blk % NCLS;
  int p = threadIdx.x;
  float v = 0.f;
  if (p < HWP) {
    const float* xc = x + (size_t)img * DD * HWP + p;
    const float* wc = w + (size_t)cls * DD;
    for (int c = 0; c < DD; ++c) v += xc[(size_t)c * HWP] * wc[c];
    v += b[cls];
    cam[(size_t)img * NCLS * HWP + (size_t)cls * HWP + p] = v;
  }
  float s = block_sum((p < HWP) ? v : 0.f, red);
  if (p == 0) logits[(size_t)img * NCLS + cls] = s / (float)HWP;
}

// ---------------- gt_aff_cam extraction ----------------
__global__ void extract_gt_aff(const float* __restrict__ cam, const int* __restrict__ label,
                               float* __restrict__ out) {
  int idx = blockIdx.x * 256 + threadIdx.x;
  if (idx >= BB * NPTS) return;
  int b = idx / NPTS, r = idx % NPTS;
  int e = r / HWP, p = r % HWP;
  out[idx] = cam[(size_t)((b * NEXO + e) * NCLS + label[b]) * HWP + p];
}

// ---------------- ego_sam ----------------
__global__ __launch_bounds__(256) void ego_sam_kernel(const float* __restrict__ attn,
                                                      float* __restrict__ out) {
  __shared__ float red[256];
  __shared__ float sam[HWP];
  int b = blockIdx.x, tid = threadIdx.x;
  if (tid < HWP) sam[tid] = 0.f;
  __syncthreads();
  const int heads[3] = {0, 1, 3};
  const float* ab = attn + (size_t)b * 6 * TOKSTRIDE * TOKSTRIDE;
#pragma unroll
  for (int hi = 0; hi < 3; ++hi) {
    int h = heads[hi];
    float a = (tid < HWP) ? ab[(size_t)h * TOKSTRIDE * TOKSTRIDE + (1 + tid)] : 0.f;
    float m = block_sum((tid < HWP) ? a : 0.f, red) / (float)HWP;
    if (tid < HWP) sam[tid] += ((a > m) ? 1.f : 0.f) * (1.f / 3.f);
    __syncthreads();
  }
  float v = (tid < HWP) ? sam[tid] : 0.f;
  float mn = block_min((tid < HWP) ? v : 3.4e38f, red);
  float mx = block_max((tid < HWP) ? v : -3.4e38f, red);
  if (tid < HWP) out[(size_t)b * HWP + tid] = (v - mn) / (mx - mn + 1e-10f);
}

// ---------------- per-sample: k-means, similarities, IoU, prototype ----------------
__global__ __launch_bounds__(256) void per_sample_kernel(
    const float* __restrict__ gt_aff, const float* __restrict__ sam_flat,
    const float* __restrict__ ego_key, const float* __restrict__ exo_key,
    float* __restrict__ sim_out, float* __restrict__ exo_sim_out, float* __restrict__ pscore_out,
    float* __restrict__ proto_ws, float* __restrict__ ok_ws) {
  __shared__ float red[256];
  __shared__ int redi[256];
  __shared__ float camn[NPTS];
  __shared__ int assign[NPTS];
  __shared__ float cen[KK * DD];
  __shared__ float sums[KK * DD];
  __shared__ float simL[KK * HWP];
  __shared__ int topidx[KK];
  const int b = blockIdx.x, tid = threadIdx.x;

  auto ptsAt = [&](int n, int c) -> float {
    int e = n / HWP, t = n % HWP;
    return exo_key[((size_t)((b * NEXO + e) * TOKSTRIDE + 1 + t)) * DD + c];
  };

  // 1. normalized CAM per exo view
  for (int e = 0; e < NEXO; ++e) {
    float v = (tid < HWP) ? gt_aff[(size_t)(b * NEXO + e) * HWP + tid] : 0.f;
    float mn = block_min((tid < HWP) ? v : 3.4e38f, red);
    float mx = block_max((tid < HWP) ? v : -3.4e38f, red);
    if (tid < HWP) camn[e * HWP + tid] = (v - mn) / (mx - mn + 1e-10f);
    __syncthreads();
  }
  // 2. weights and validity
  float wsl = 0.f;
  for (int n = tid; n < NPTS; n += 256) wsl += (camn[n] > 0.6f) ? 1.f : 0.f;
  float wsum = block_sum(wsl, red);
  bool valid = (wsum >= (float)KK);
  // 3. top-K of camn*wts
  for (int k = 0; k < KK; ++k) {
    float bestv = -1e30f; int besti = 0x7fffffff;
    for (int n = tid; n < NPTS; n += 256) {
      bool taken = false;
      for (int j = 0; j < k; ++j) if (topidx[j] == n) taken = true;
      float s = (camn[n] > 0.6f) ? camn[n] : 0.f;
      if (!taken && (s > bestv || (s == bestv && n < besti))) { bestv = s; besti = n; }
    }
    int gi = block_argmax(bestv, besti, red, redi);
    if (tid == 0) topidx[k] = gi;
    __syncthreads();
  }
  // 4. init centroids
  for (int j = tid; j < KK * DD; j += 256) cen[j] = ptsAt(topidx[j / DD], j % DD);
  __syncthreads();
  // 5. k-means (20 iters)
  for (int it = 0; it < 20; ++it) {
    float cen2[KK];
    for (int k = 0; k < KK; ++k) {
      float s = 0.f;
      for (int c = tid; c < DD; c += 256) s += cen[k * DD + c] * cen[k * DD + c];
      cen2[k] = block_sum(s, red);
    }
    for (int n = tid; n < NPTS; n += 256) {
      float d0 = 0.f, d1 = 0.f, d2 = 0.f;
      for (int c = 0; c < DD; ++c) {
        float x = ptsAt(n, c);
        d0 += x * cen[c]; d1 += x * cen[DD + c]; d2 += x * cen[2 * DD + c];
      }
      float s0 = cen2[0] - 2.f * d0, s1 = cen2[1] - 2.f * d1, s2 = cen2[2] - 2.f * d2;
      int a = 0; float sb = s0;
      if (s1 < sb) { sb = s1; a = 1; }
      if (s2 < sb) { sb = s2; a = 2; }
      assign[n] = a;
    }
    __syncthreads();
    float cnt[KK];
    for (int k = 0; k < KK; ++k) {
      float s = 0.f;
      for (int n = tid; n < NPTS; n += 256) s += (assign[n] == k && camn[n] > 0.6f) ? 1.f : 0.f;
      cnt[k] = block_sum(s, red);
    }
    for (int j = tid; j < KK * DD; j += 256) {
      int k = j / DD, c = j % DD;
      float s = 0.f;
      for (int n = 0; n < NPTS; ++n)
        if (assign[n] == k && camn[n] > 0.6f) s += ptsAt(n, c);
      sums[j] = s;
    }
    __syncthreads();
    for (int j = tid; j < KK * DD; j += 256) {
      int k = j / DD;
      if (cnt[k] > 0.f) cen[j] = sums[j] / fmaxf(cnt[k], 1.f);
    }
    __syncthreads();
  }
  // 6. normalize clusters
  for (int k = 0; k < KK; ++k) {
    float s = 0.f;
    for (int c = tid; c < DD; c += 256) s += cen[k * DD + c] * cen[k * DD + c];
    float rn = 1.f / fmaxf(sqrtf(block_sum(s, red)), 1e-12f);
    for (int c = tid; c < DD; c += 256) cen[k * DD + c] *= rn;
    __syncthreads();
  }
  float vf = valid ? 1.f : 0.f;
  // 7. exo similarity maps
  for (int n = tid; n < NPTS; n += 256) {
    float nn = 0.f, d0 = 0.f, d1 = 0.f, d2 = 0.f;
    for (int c = 0; c < DD; ++c) {
      float x = ptsAt(n, c);
      nn += x * x; d0 += x * cen[c]; d1 += x * cen[DD + c]; d2 += x * cen[2 * DD + c];
    }
    float rn = 1.f / fmaxf(sqrtf(nn), 1e-12f);
    int e = n / HWP, t = n % HWP;
    size_t base = ((size_t)(b * NEXO + e) * KK) * HWP + t;
    exo_sim_out[base + 0 * HWP] = d0 * rn * vf;
    exo_sim_out[base + 1 * HWP] = d1 * rn * vf;
    exo_sim_out[base + 2 * HWP] = d2 * rn * vf;
  }
  // 8. ego similarity maps
  for (int t = tid; t < HWP; t += 256) {
    float nn = 0.f, d0 = 0.f, d1 = 0.f, d2 = 0.f;
    const float* row = ego_key + ((size_t)(b * TOKSTRIDE + 1 + t)) * DD;
    for (int c = 0; c < DD; ++c) {
      float x = row[c];
      nn += x * x; d0 += x * cen[c]; d1 += x * cen[DD + c]; d2 += x * cen[2 * DD + c];
    }
    float rn = 1.f / fmaxf(sqrtf(nn), 1e-12f);
    simL[t] = d0 * rn; simL[HWP + t] = d1 * rn; simL[2 * HWP + t] = d2 * rn;
    sim_out[((size_t)(b * KK + 0)) * HWP + t] = d0 * rn * vf;
    sim_out[((size_t)(b * KK + 1)) * HWP + t] = d1 * rn * vf;
    sim_out[((size_t)(b * KK + 2)) * HWP + t] = d2 * rn * vf;
  }
  __syncthreads();
  // 9. IoU-like scores
  float sv = (tid < HWP) ? sam_flat[(size_t)b * HWP + tid] : 0.f;
  float sam_mean = block_sum((tid < HWP) ? sv : 0.f, red) / (float)HWP;
  float samh = (tid < HWP && sv > sam_mean) ? 1.f : 0.f;
  float sam_sum = block_sum(samh, red);
  float pscore[KK];
  for (int k = 0; k < KK; ++k) {
    float s = (tid < HWP) ? simL[k * HWP + tid] : 0.f;
    float mn = block_min((tid < HWP) ? s : 3.4e38f, red);
    float mx = block_max((tid < HWP) ? s : -3.4e38f, red);
    float sn = (s - mn) / (mx - mn + 1e-12f);
    float thr = block_sum((tid < HWP) ? sn : 0.f, red) / (float)HWP;
    float hard = (tid < HWP && sn > thr) ? 1.f : 0.f;
    float inter = block_sum(hard * samh, red);
    float shs = block_sum(hard, red);
    float uni = shs + sam_sum - inter;
    pscore[k] = (inter / shs + sam_sum / uni) * 0.5f;
  }
  float pmax = fmaxf(pscore[0], fmaxf(pscore[1], pscore[2]));
  float po0 = pscore[0] * vf, po1 = pscore[1] * vf, po2 = pscore[2] * vf;
  int kb = 0; float pb = po0;
  if (po1 > pb) { pb = po1; kb = 1; }
  if (po2 > pb) { pb = po2; kb = 2; }
  bool ok = valid && (pmax >= 0.6f);
  if (tid == 0) {
    pscore_out[b * KK + 0] = po0;
    pscore_out[b * KK + 1] = po1;
    pscore_out[b * KK + 2] = po2;
    ok_ws[b] = ok ? 1.f : 0.f;
  }
  for (int c = tid; c < DD; c += 256)
    proto_ws[(size_t)b * DD + c] = ok ? cen[kb * DD + c] : 0.f;
}

// ---------------- gt_ego_cam (select label channel, minmax) ----------------
__global__ __launch_bounds__(256) void gt_ego_cam_kernel(const float* __restrict__ pred,
                                                         const int* __restrict__ label,
                                                         float* __restrict__ out) {
  __shared__ float red[256];
  int b = blockIdx.x, tid = threadIdx.x;
  int L = label[b];
  float v = (tid < HWP) ? pred[((size_t)b * NCLS + L) * HWP + tid] : 0.f;
  float mn = block_min((tid < HWP) ? v : 3.4e38f, red);
  float mx = block_max((tid < HWP) ? v : -3.4e38f, red);
  if (tid < HWP) out[(size_t)b * HWP + tid] = (v - mn) / (mx - mn + 1e-10f);
}

// ---------------- concentration loss (per sample) ----------------
__global__ __launch_bounds__(256) void loss_con_kernel(const float* __restrict__ pred,
                                                       float* __restrict__ lcon_ws) {
  __shared__ float red[256];
  int b = blockIdx.x, tid = threadIdx.x;
  float ys = (float)(tid / WW) * (1.f / (float)HH);
  float xs = (float)(tid % WW) * (1.f / (float)WW);
  float acc = 0.f;
  for (int cls = 0; cls < NCLS; ++cls) {
    float l = (tid < HWP) ? pred[((size_t)b * NCLS + cls) * HWP + tid] : -3.4e38f;
    float mx = block_max(l, red);
    float e = (tid < HWP) ? expf(l - mx) : 0.f;
    float se = block_sum(e, red);
    float p = e / se;
    float muy = block_sum(p * ys, red);
    float mux = block_sum(p * xs, red);
    float dy = ys - muy, dx = xs - mux;
    acc += block_sum(p * (dy * dy + dx * dx), red);
  }
  if (tid == 0) lcon_ws[b] = acc / (float)NCLS;
}

// ---------------- prototype loss (per sample) ----------------
__global__ __launch_bounds__(256) void loss_proto_kernel(
    const float* __restrict__ ego_key, const float* __restrict__ gcam,
    const float* __restrict__ proto_ws, const float* __restrict__ ok_ws,
    float* __restrict__ lp_ws) {
  __shared__ float red[256];
  __shared__ float emb[DD];
  int b = blockIdx.x, tid = threadIdx.x;
  float g = (tid < HWP) ? gcam[(size_t)b * HWP + tid] : 0.f;
  float den = block_sum(g, red);
  for (int c = tid; c < DD; c += 256) {
    float s = 0.f;
    for (int p = 0; p < HWP; ++p)
      s += ego_key[((size_t)(b * TOKSTRIDE + 1 + p)) * DD + c] * gcam[(size_t)b * HWP + p];
    emb[c] = s / den;
  }
  __syncthreads();
  float d = 0.f, e2 = 0.f, p2 = 0.f;
  for (int c = tid; c < DD; c += 256) {
    float ev = emb[c], pv = proto_ws[(size_t)b * DD + c];
    d += ev * pv; e2 += ev * ev; p2 += pv * pv;
  }
  d = block_sum(d, red);
  e2 = block_sum(e2, red);
  p2 = block_sum(p2, red);
  if (tid == 0) {
    float cosv = d / (fmaxf(sqrtf(e2), 1e-8f) * fmaxf(sqrtf(p2), 1e-8f));
    lp_ws[b] = fmaxf(1.f - cosv - 0.5f, 0.f) * ok_ws[b];
  }
}

// ---------------- final loss reduction ----------------
__global__ __launch_bounds__(256) void final_losses(const float* __restrict__ lcon_ws,
                                                    const float* __restrict__ lp_ws,
                                                    const float* __restrict__ ok_ws,
                                                    const int* __restrict__ ep0,
                                                    const int* __restrict__ ep1,
                                                    float* __restrict__ loss_proto_out,
                                                    float* __restrict__ loss_con_out) {
  __shared__ float red[256];
  int tid = threadIdx.x;
  float a = (tid < BB) ? lcon_ws[tid] : 0.f;
  float p = (tid < BB) ? lp_ws[tid] : 0.f;
  float o = (tid < BB) ? ok_ws[tid] : 0.f;
  float sa = block_sum(a, red);
  float sp = block_sum(p, red);
  float so = block_sum(o, red);
  if (tid == 0) {
    *loss_con_out = sa / (float)BB;
    float flag = (ep0[0] > ep1[0]) ? 1.f : 0.f;
    *loss_proto_out = flag * (sp / (so + 1e-15f));
  }
}

// ================== host-side launch ==================
static inline int cdiv(long long a, long long b) { return (int)((a + b - 1) / b); }

extern "C" void kernel_launch(void* const* d_in, const int* in_sizes, int n_in,
                              void* d_out, int out_size, void* d_ws, size_t ws_size,
                              hipStream_t stream) {
  (void)in_sizes; (void)n_in; (void)out_size; (void)ws_size;
  // inputs: setup_inputs order, params flattened in sorted-key order (JAX pytree)
  const float* ego_key  = (const float*)d_in[0];
  const float* exo_key  = (const float*)d_in[1];
  const float* ego_attn = (const float*)d_in[2];
  const int*   label    = (const int*)d_in[3];
  const int*   ep0      = (const int*)d_in[4];
  const int*   ep1      = (const int*)d_in[5];
  const float* ego_b1  = (const float*)d_in[6];
  const float* ego_b2  = (const float*)d_in[7];
  const float* ego_bt1 = (const float*)d_in[8];
  const float* ego_bt2 = (const float*)d_in[9];
  const float* ego_g1  = (const float*)d_in[10];
  const float* ego_g2  = (const float*)d_in[11];
  const float* ego_w1  = (const float*)d_in[12];
  const float* ego_w2  = (const float*)d_in[13];
  const float* exo_b1  = (const float*)d_in[14];
  const float* exo_b2  = (const float*)d_in[15];
  const float* exo_bt1 = (const float*)d_in[16];
  const float* exo_bt2 = (const float*)d_in[17];
  const float* exo_g1  = (const float*)d_in[18];
  const float* exo_g2  = (const float*)d_in[19];
  const float* exo_w1  = (const float*)d_in[20];
  const float* exo_w2  = (const float*)d_in[21];
  const float* fc1_b   = (const float*)d_in[22];
  const float* fc1_w   = (const float*)d_in[23];
  const float* fc2_b   = (const float*)d_in[24];
  const float* fc2_w   = (const float*)d_in[25];
  const float* fc_b    = (const float*)d_in[26];
  const float* fc_w    = (const float*)d_in[27];
  const float* ln_b    = (const float*)d_in[28];
  const float* ln_g    = (const float*)d_in[29];

  // workspace carve-out
  size_t cur = 0;
  char* base = (char*)d_ws;
  auto alloc = [&](size_t bytes) -> char* {
    char* p = base + cur;
    cur += (bytes + 255) & ~(size_t)255;
    return p;
  };
  _Float16* w1h   = (_Float16*)alloc((size_t)D4 * DD * 2);
  _Float16* w2h   = (_Float16*)alloc((size_t)DD * D4 * 2);
  _Float16* wce1  = (_Float16*)alloc((size_t)DD * DD * 9 * 2);
  _Float16* wce2  = (_Float16*)alloc((size_t)DD * DD * 9 * 2);
  _Float16* wcx1  = (_Float16*)alloc((size_t)DD * DD * 9 * 2);
  _Float16* wcx2  = (_Float16*)alloc((size_t)DD * DD * 9 * 2);
  float* ln_ego   = (float*)alloc((size_t)BB * HWP * 2 * 4);
  float* ln_exo   = (float*)alloc((size_t)NI * HWP * 2 * 4);
  _Float16* ego_hwc = (_Float16*)alloc((size_t)BB * HWP * DD * 2);
  _Float16* exo_hwc = (_Float16*)alloc((size_t)NI * HWP * DD * 2);
  float* cb1      = (float*)alloc((size_t)NI * DD * HWP * 4);
  float* cb2      = (float*)alloc((size_t)NI * DD * HWP * 4);
  _Float16* hwc2  = (_Float16*)alloc((size_t)NI * HWP * DD * 2);
  float* bnmu     = (float*)alloc(DD * 4);
  float* bnrs     = (float*)alloc(DD * 4);
  float* affcam   = (float*)alloc((size_t)NI * NCLS * HWP * 4);
  float* egopred  = (float*)alloc((size_t)BB * NCLS * HWP * 4);
  float* proto_ws = (float*)alloc((size_t)BB * DD * 4);
  float* ok_ws    = (float*)alloc((size_t)BB * 4);
  float* lcon_ws  = (float*)alloc((size_t)BB * 4);
  float* lp_ws    = (float*)alloc((size_t)BB * 4);

  // output sections (reference return order)
  float* out = (float*)d_out;
  float* o_gtaff  = out + 0;        // [64,3,196]
  float* o_sam    = out + 37632;    // [64,196]
  float* o_sim    = out + 50176;    // [64,3,196]
  float* o_exosim = out + 87808;    // [64,3,3,196]
  float* o_pscore = out + 200704;   // [64,3]
  float* o_gtego  = out + 200896;   // [64,196]
  float* o_logits = out + 213440;   // [64,3,36]
  float* o_logego = out + 220352;   // [64,36]
  float* o_lproto = out + 222656;   // scalar
  float* o_lcon   = out + 222657;   // scalar

  const int ntok_ego = BB * HWP, ntok_exo = NI * HWP;

  // weights -> f16
  cvt_f16<<<cdiv(D4 * DD, 256), 256, 0, stream>>>(fc1_w, w1h, D4 * DD);
  cvt_f16<<<cdiv(DD * D4, 256), 256, 0, stream>>>(fc2_w, w2h, DD * D4);
  cvt_convw<<<cdiv(DD * DD * 9, 256), 256, 0, stream>>>(ego_w1, wce1);
  cvt_convw<<<cdiv(DD * DD * 9, 256), 256, 0, stream>>>(ego_w2, wce2);
  cvt_convw<<<cdiv(DD * DD * 9, 256), 256, 0, stream>>>(exo_w1, wcx1);
  cvt_convw<<<cdiv(DD * DD * 9, 256), 256, 0, stream>>>(exo_w2, wcx2);

  // LN stats + fused MLP
  ln_stats<<<cdiv(ntok_ego, 256), 256, 0, stream>>>(ego_key, ln_ego, ntok_ego);
  ln_stats<<<cdiv(ntok_exo, 256), 256, 0, stream>>>(exo_key, ln_exo, ntok_exo);
  mlp_fused<<<ntok_ego / 16, 256, 0, stream>>>(ego_key, ln_g, ln_b, ln_ego, w1h, fc1_b, w2h, fc2_b, ego_hwc, ntok_ego);
  mlp_fused<<<ntok_exo / 16, 256, 0, stream>>>(exo_key, ln_g, ln_b, ln_exo, w1h, fc1_b, w2h, fc2_b, exo_hwc, ntok_exo);

  // exo conv block
  conv3x3_wmma<<<dim3(39, NI), 256, 0, stream>>>(exo_hwc, wcx1, exo_b1, cb1);
  bn_stats<<<DD, 256, 0, stream>>>(cb1, NI, bnmu, bnrs);
  bn_relu_f16hwc<<<cdiv((long long)NI * DD * HWP, 256), 256, 0, stream>>>(cb1, bnmu, bnrs, exo_g1, exo_bt1, hwc2, NI);
  conv3x3_wmma<<<dim3(39, NI), 256, 0, stream>>>(hwc2, wcx2, exo_b2, cb2);
  bn_stats<<<DD, 256, 0, stream>>>(cb2, NI, bnmu, bnrs);
  bn_relu_f32<<<cdiv((long long)NI * DD * HWP, 256), 256, 0, stream>>>(cb2, bnmu, bnrs, exo_g2, exo_bt2, cb1, NI);
  fc_conv<<<NI * NCLS, 256, 0, stream>>>(cb1, fc_w, fc_b, affcam, o_logits);

  // gt_aff_cam, ego_sam, per-sample pipeline
  extract_gt_aff<<<cdiv(BB * NPTS, 256), 256, 0, stream>>>(affcam, label, o_gtaff);
  ego_sam_kernel<<<BB, 256, 0, stream>>>(ego_attn, o_sam);
  per_sample_kernel<<<BB, 256, 0, stream>>>(o_gtaff, o_sam, ego_key, exo_key,
                                            o_sim, o_exosim, o_pscore, proto_ws, ok_ws);

  // ego conv block (reuses cb1/cb2/hwc2)
  conv3x3_wmma<<<dim3(39, BB), 256, 0, stream>>>(ego_hwc, wce1, ego_b1, cb1);
  bn_stats<<<DD, 256, 0, stream>>>(cb1, BB, bnmu, bnrs);
  bn_relu_f16hwc<<<cdiv((long long)BB * DD * HWP, 256), 256, 0, stream>>>(cb1, bnmu, bnrs, ego_g1, ego_bt1, hwc2, BB);
  conv3x3_wmma<<<dim3(39, BB), 256, 0, stream>>>(hwc2, wce2, ego_b2, cb2);
  bn_stats<<<DD, 256, 0, stream>>>(cb2, BB, bnmu, bnrs);
  bn_relu_f32<<<cdiv((long long)BB * DD * HWP, 256), 256, 0, stream>>>(cb2, bnmu, bnrs, ego_g2, ego_bt2, cb1, BB);
  fc_conv<<<BB * NCLS, 256, 0, stream>>>(cb1, fc_w, fc_b, egopred, o_logego);

  // ego cam + losses
  gt_ego_cam_kernel<<<BB, 256, 0, stream>>>(egopred, label, o_gtego);
  loss_con_kernel<<<BB, 256, 0, stream>>>(egopred, lcon_ws);
  loss_proto_kernel<<<BB, 256, 0, stream>>>(ego_key, o_gtego, proto_ws, ok_ws, lp_ws);
  final_losses<<<1, 256, 0, stream>>>(lcon_ws, lp_ws, ok_ws, ep0, ep1, o_lproto, o_lcon);
}
